// RecurrentEncoder_53609781789068
// MI455X (gfx1250) — matrix operand order repaired and checked by reading, not compiled, not run on hardware
//
#include <hip/hip_runtime.h>
#include <hip/hip_bf16.h>
#include <math.h>

// Problem constants (match reference)
#define BN 4096
#define DN 256
#define HN 512
#define KN 64
#define TN 32

typedef __attribute__((ext_vector_type(16))) _Float16 v16h;
typedef __attribute__((ext_vector_type(8)))  _Float16 v8h;
typedef __attribute__((ext_vector_type(8)))  float    v8f;

// ---- WMMA helpers (CDNA5 16x16x32 f16 -> f32) -------------------------------

__device__ __forceinline__ v8f wmma16(v16h a, v16h b, v8f c) {
  // (neg_a, A, neg_b, B, c_mod, C, reuse_a, reuse_b)
  return __builtin_amdgcn_wmma_f32_16x16x32_f16(false, a, false, b, (short)0, c,
                                                false, false);
}

// A-matrix 16x32 f16 fragment from a row-major f16 tile with row stride 512.
// lane l16 holds row M=l16; half kh selects K range [16kh,16kh+16)
__device__ __forceinline__ v16h ldA512(const _Float16* base, int l16, int k0, int kh) {
  const v8h c0 = *(const v8h*)(base + l16 * 512 + k0 + 16 * kh);
  const v8h c1 = *(const v8h*)(base + l16 * 512 + k0 + 16 * kh + 8);
  v16h a;
#pragma unroll
  for (int i = 0; i < 8; ++i) { a[i] = c0[i]; a[i + 8] = c1[i]; }
  return a;
}

// B-matrix 32x16 f16 fragment: B(k,n) = W[n][k]; lane l16 -> column n, half kh
// selects K range [k0+16kh, +16). Row-major W, leading dim ld (contiguous 32B).
__device__ __forceinline__ v16h ldB(const _Float16* W, int ld, int n, int k0, int kh) {
  return *(const v16h*)(W + (size_t)n * ld + k0 + 16 * kh);
}

// ---- Kernel 0: fp32 -> f16 weight conversion --------------------------------

__global__ __launch_bounds__(256) void convert_weights_kernel(
    const float* __restrict__ Wih, const float* __restrict__ Whh,
    const float* __restrict__ Wfe, _Float16* __restrict__ oWih,
    _Float16* __restrict__ oWhh, _Float16* __restrict__ oWfe) {
  int i = blockIdx.x * 256 + threadIdx.x;            // grid covers 512*512
  if (i < HN * DN)     oWih[i] = (_Float16)Wih[i];
  if (i < HN * HN)     oWhh[i] = (_Float16)Whh[i];
  if (i < 2 * KN * HN) oWfe[i] = (_Float16)Wfe[i];
}

// ---- Kernel 1: xw = x @ W_ih^T + b_ih  (B x H, fp32 out) --------------------

__global__ __launch_bounds__(256) void xw_kernel(
    const float* __restrict__ x,        // B x D fp32
    const _Float16* __restrict__ Wih,   // H x D f16
    const float* __restrict__ b_ih,     // H
    float* __restrict__ xw) {           // B x H fp32
  const int tid = threadIdx.x;
  const int wav = tid >> 5;
  const int lane = tid & 31;
  const int l16 = lane & 15;
  const int kh  = lane >> 4;
  const int r0 = blockIdx.x * 16;       // batch rows of this WG
  const int n0 = wav * 64;              // output cols of this wave

  v8f acc[4];
#pragma unroll
  for (int nt = 0; nt < 4; ++nt)
#pragma unroll
    for (int r = 0; r < 8; ++r) acc[nt][r] = 0.0f;

  const float* xp = x + (size_t)(r0 + l16) * DN;
  for (int k0 = 0; k0 < DN; k0 += 32) {
    v16h a;
#pragma unroll
    for (int i = 0; i < 8; ++i) {
      a[i]     = (_Float16)xp[k0 + 16 * kh + i];
      a[i + 8] = (_Float16)xp[k0 + 16 * kh + 8 + i];
    }
#pragma unroll
    for (int nt = 0; nt < 4; ++nt) {
      v16h b = ldB(Wih, DN, n0 + 16 * nt + l16, k0, kh);
      acc[nt] = wmma16(a, b, acc[nt]);
    }
  }
#pragma unroll
  for (int nt = 0; nt < 4; ++nt) {
    int n = n0 + 16 * nt + l16;
    float bi = b_ih[n];
#pragma unroll
    for (int r = 0; r < 8; ++r) {
      int m = r + 8 * kh;
      xw[(size_t)(r0 + m) * HN + n] = acc[nt][r] + bi;
    }
  }
}

// ---- Kernel 2: persistent recurrence ---------------------------------------
// Each WG owns 32 batch rows for all 32 ticks; hx lives in LDS as f16.
// Each wave owns a 64-col N-range for ALL 32 rows -> each W_hh element is read
// exactly once per WG per tick (no intra-WG duplication).

__global__ __launch_bounds__(256) void rnn_kernel(
    const float* __restrict__ xw,       // B x H fp32 (workspace)
    const _Float16* __restrict__ Whh,   // H x H f16
    const _Float16* __restrict__ Wfe,   // 2K x H f16
    const float* __restrict__ b_hh,     // H
    const float* __restrict__ b_fe,     // 2K
    const float* __restrict__ eps,      // T x B x K fp32
    float* __restrict__ out) {          // zs | mus | sigs | hx (fp32)
  __shared__ __attribute__((aligned(32))) _Float16 hx_sh[32 * HN];     // 32 KB
  __shared__ __attribute__((aligned(32))) float    xw_sh[32 * HN];     // 64 KB
  __shared__ __attribute__((aligned(32))) float    fe_sh[32 * 2 * KN]; // 16 KB

  const int tid = threadIdx.x;
  const int wav = tid >> 5;
  const int lane = tid & 31;
  const int l16 = lane & 15;
  const int kh  = lane >> 4;
  const int r0 = blockIdx.x * 32;       // batch rows of this WG
  const int n0 = wav * 64;              // H cols of this wave (main GEMM)
  const int fc0 = wav * 16;             // fe cols of this wave (8 waves = 128)

  // Stage xw tile via async global->LDS copies (CDNA5 async path); zero hx.
  {
    const unsigned lds_base = (unsigned)(uintptr_t)xw_sh;
    for (int j = tid; j < 32 * HN / 4; j += 256) {   // 4096 x 16B vectors
      int row = j >> 7;                // 128 vectors per 512-float row
      int c   = (j & 127) * 4;
      const float* g = xw + (size_t)(r0 + row) * HN + c;
      unsigned l = lds_base + (unsigned)j * 16u;
      asm volatile("global_load_async_to_lds_b128 %0, %1, off"
                   :: "v"(l), "v"(g) : "memory");
    }
    uint4 z4 = {0u, 0u, 0u, 0u};
    for (int j = tid; j < 32 * HN / 8; j += 256)     // zero hx (f16) as 16B
      ((uint4*)hx_sh)[j] = z4;
    asm volatile("s_wait_asynccnt 0x0" ::: "memory");
  }

  // Preload per-lane biases.
  float bh[4];
#pragma unroll
  for (int nt = 0; nt < 4; ++nt) bh[nt] = b_hh[n0 + 16 * nt + l16];
  const float bf = b_fe[fc0 + l16];
  __syncthreads();

  const size_t ZN = (size_t)BN * KN * TN;
  float* out_z = out;
  float* out_m = out + ZN;
  float* out_s = out + 2 * ZN;
  float* out_h = out + 3 * ZN;

  for (int t = 0; t < TN; ++t) {
    // ---- acc = xw + hx @ W_hh^T (wave: 32 rows x 64 cols, 8 tiles) ----
    v8f acc[2][4];
#pragma unroll
    for (int g = 0; g < 2; ++g)
#pragma unroll
      for (int nt = 0; nt < 4; ++nt)
#pragma unroll
        for (int r = 0; r < 8; ++r)
          acc[g][nt][r] = xw_sh[(16 * g + r + 8 * kh) * HN + n0 + 16 * nt + l16];

    // Software-pipelined B fragments (W_hh streamed from L2).
    v16h bcur[4];
#pragma unroll
    for (int nt = 0; nt < 4; ++nt) bcur[nt] = ldB(Whh, HN, n0 + 16 * nt + l16, 0, kh);

    for (int k0 = 0; k0 < HN; k0 += 32) {
      const int kn = (k0 + 32) & (HN - 1);   // wraps to 0 on last iter (unused)
      v16h bnext[4];
#pragma unroll
      for (int nt = 0; nt < 4; ++nt)
        bnext[nt] = ldB(Whh, HN, n0 + 16 * nt + l16, kn, kh);
      v16h a0 = ldA512(hx_sh, l16, k0, kh);
      v16h a1 = ldA512(hx_sh + 16 * HN, l16, k0, kh);
#pragma unroll
      for (int nt = 0; nt < 4; ++nt) {
        acc[0][nt] = wmma16(a0, bcur[nt], acc[0][nt]);
        acc[1][nt] = wmma16(a1, bcur[nt], acc[1][nt]);
      }
#pragma unroll
      for (int nt = 0; nt < 4; ++nt) bcur[nt] = bnext[nt];
    }
    __syncthreads();  // B1: all reads of old hx complete

    // ---- hx = tanh(acc + b_hh); write f16 back to LDS ----
#pragma unroll
    for (int g = 0; g < 2; ++g) {
#pragma unroll
      for (int nt = 0; nt < 4; ++nt) {
#pragma unroll
        for (int r = 0; r < 8; ++r) {
          int m = 16 * g + r + 8 * kh;
          int n = n0 + 16 * nt + l16;
          float h = tanhf(acc[g][nt][r] + bh[nt]);
          hx_sh[m * HN + n] = (_Float16)h;
          if (t == TN - 1)
            __builtin_nontemporal_store(h, out_h + (size_t)(r0 + m) * HN + n);
        }
      }
    }
    __syncthreads();  // B2: new hx visible to everyone

    // ---- fe = hx @ W_fe^T: each wave a 16-col slab for BOTH M-groups ----
    v8f fa[2];
#pragma unroll
    for (int g = 0; g < 2; ++g)
#pragma unroll
      for (int r = 0; r < 8; ++r) fa[g][r] = 0.0f;
    const _Float16* WfeRow = Wfe + (size_t)(fc0 + l16) * HN;
    v16h fbcur = *(const v16h*)(WfeRow + 16 * kh);
    for (int k0 = 0; k0 < HN; k0 += 32) {
      const int kn = (k0 + 32) & (HN - 1);
      v16h fbnext = *(const v16h*)(WfeRow + kn + 16 * kh);
      v16h a0 = ldA512(hx_sh, l16, k0, kh);
      v16h a1 = ldA512(hx_sh + 16 * HN, l16, k0, kh);
      fa[0] = wmma16(a0, fbcur, fa[0]);    // B fragment reused for both Ms
      fa[1] = wmma16(a1, fbcur, fa[1]);
      fbcur = fbnext;
    }
#pragma unroll
    for (int g = 0; g < 2; ++g) {
#pragma unroll
      for (int r = 0; r < 8; ++r) {
        int m = 16 * g + r + 8 * kh;
        float v = fa[g][r] + bf;
        if (wav >= 4) {  // sig half: softplus(fe - 5)
          float u = v - 5.0f;
          v = (u > 20.0f) ? u : log1pf(expf(u));
        }
        fe_sh[m * 128 + fc0 + l16] = v;
      }
    }
    __syncthreads();  // B3: fe slab complete in LDS

    // ---- z = eps * sig + mus; out in (B,K,T) layout. REGULAR stores: each
    // 128B output line accumulates one dword per tick and must stay resident
    // in L2 (100 MB of lines << 192 MB L2) to write back to HBM only once.
    for (int i = tid; i < 32 * KN; i += 256) {
      int m = i >> 6, k = i & 63;
      float mu = fe_sh[m * 128 + k];
      float sg = fe_sh[m * 128 + 64 + k];
      float e  = __builtin_nontemporal_load(
                     eps + (size_t)t * BN * KN + (size_t)(r0 + m) * KN + k);
      size_t o = (size_t)(r0 + m) * KN * TN + (size_t)k * TN + t;
      out_z[o] = e * sg + mu;
      out_m[o] = mu;
      out_s[o] = sg;
    }
    // No extra barrier: next tick's LDS writes are behind B1/B2.
  }
}

// ---- Launch -----------------------------------------------------------------

extern "C" void kernel_launch(void* const* d_in, const int* in_sizes, int n_in,
                              void* d_out, int out_size, void* d_ws, size_t ws_size,
                              hipStream_t stream) {
  (void)in_sizes; (void)n_in; (void)out_size; (void)ws_size;
  const float* x    = (const float*)d_in[0];   // B x D
  const float* eps  = (const float*)d_in[1];   // T x B x K
  const float* Wih  = (const float*)d_in[2];   // H x D
  const float* Whh  = (const float*)d_in[3];   // H x H
  const float* b_ih = (const float*)d_in[4];   // H
  const float* b_hh = (const float*)d_in[5];   // H
  const float* Wfe  = (const float*)d_in[6];   // 2K x H
  const float* b_fe = (const float*)d_in[7];   // 2K
  float* out = (float*)d_out;

  // Workspace layout (bytes): xw fp32 | Wih f16 | Whh f16 | Wfe f16
  char* ws = (char*)d_ws;
  float*    xw    = (float*)ws;                         // 4096*512*4 = 8388608
  _Float16* Wih_h = (_Float16*)(ws + 8388608);          // 512*256*2 = 262144
  _Float16* Whh_h = (_Float16*)(ws + 8388608 + 262144); // 512*512*2 = 524288
  _Float16* Wfe_h = (_Float16*)(ws + 8388608 + 262144 + 524288); // 128*512*2

  convert_weights_kernel<<<(HN * HN) / 256, 256, 0, stream>>>(
      Wih, Whh, Wfe, Wih_h, Whh_h, Wfe_h);
  xw_kernel<<<BN / 16, 256, 0, stream>>>(x, Wih_h, b_ih, xw);
  rnn_kernel<<<BN / 32, 256, 0, stream>>>(xw, Whh_h, Wfe_h, b_hh, b_fe, eps, out);
}
